// Model_33749853012505
// MI455X (gfx1250) — matrix operand order repaired and checked by reading, not compiled, not run on hardware
//
#include <hip/hip_runtime.h>
#include <cstdint>
#include <cstddef>

// ---------------- model constants ----------------
#define SEQ_LEN   512
#define PRED_LEN  96
#define PATCH_LEN 16
#define STRIDE_   8
#define D_MODEL   128
#define D_FF      256
#define D_STATE   16
#define N_VARS    32
#define BATCH     64
#define PATCH_NUM 64          // (512-16)/8 + 1 + 1
#define D_INNER   256
#define DT_RANK   8
#define T_TOK     131072      // BATCH * N_VARS * PATCH_NUM == BATCH * PATCH_NUM * N_VARS

#define CDIV(a,b) (((a)+(b)-1)/(b))

// ---------------- WMMA types ----------------
typedef __attribute__((ext_vector_type(16))) __bf16 v16bf;
typedef __attribute__((ext_vector_type(8)))  float  v8f;

union FragBF { v16bf v; uint4 q[2]; };

// activation codes
#define ACT_NONE     0
#define ACT_SILU     1
#define ACT_GELU     2
#define ACT_SOFTPLUS 3

__device__ __forceinline__ float act_silu(float x) { return x / (1.f + __expf(-x)); }
__device__ __forceinline__ float act_gelu(float x) {
  float x3 = x * x * x;
  return 0.5f * x * (1.f + tanhf(0.7978845608f * (x + 0.044715f * x3)));
}
__device__ __forceinline__ float act_softplus(float x) {
  return fmaxf(x, 0.f) + log1pf(__expf(-fabsf(x)));
}
template <int ACT>
__device__ __forceinline__ float apply_act(float v) {
  if (ACT == ACT_SILU) return act_silu(v);
  if (ACT == ACT_GELU) return act_gelu(v);
  if (ACT == ACT_SOFTPLUS) return act_softplus(v);
  return v;
}

// ---- optional gfx1250 async global->LDS staging (guarded; falls back) ----
#if defined(__has_builtin)
#  if __has_builtin(__builtin_amdgcn_global_load_async_to_lds_b128)
#    define ASYNC_LDS 1
#  endif
#endif

#ifdef ASYNC_LDS
// types matching the builtin's parameters exactly (per clang diagnostics):
//   param0: int __attribute__((vector_size(16))) addrspace(1)*   (non-const)
//   param1: int __attribute__((vector_size(16))) addrspace(3)*
typedef int v4i_vs __attribute__((vector_size(16)));
typedef __attribute__((address_space(1))) v4i_vs* as1_v4i_t;
typedef __attribute__((address_space(3))) v4i_vs* as3_v4i_t;
#endif

// =====================================================================
// Tiled GEMM: C[M,N] = act(A[M,K] @ B[K,N] + bias)
//   - B pre-transposed & bf16-converted: Bt[Npad][Kpad], Npad mult of 64
//   - workgroup = 4 waves; output tile 64(M) x 64(N): each wave owns a
//     16-row stripe and 4 accumulators (4 x v_wmma_f32_16x16x32_bf16 / k-step)
//   CONTRACT (all call sites satisfy it):
//     * M % 64 == 0
//     * every A row is readable for ceil32(K) columns (lda >= ceil32(K))
//   Optional split-K: grid.z = S chunks of kChunk; partial sums written to
//   C + z*cStride with no bias/act (reduced by reduce_sk, fixed order).
// =====================================================================
#define LDSROW 40   // 32 bf16 data + 8 pad -> conflict-free b128 gathers
#define NT 4        // N tiles per workgroup

template <int ACT>
__global__ __launch_bounds__(128) void gemm_wmma_bf16(
    const float* __restrict__ A, int lda,
    const __bf16* __restrict__ Bt, int Kpad,
    const float* __restrict__ bias,
    float* __restrict__ C, int ldc,
    int N, int K, int kChunk, size_t cStride)
{
  __shared__ __bf16 As[64 * LDSROW];
  __shared__ __bf16 Bs[64 * LDSROW];

  const int tid  = threadIdx.x;
  const int lane = tid & 31;
  const int wave = tid >> 5;
  const int m0   = blockIdx.x * 64;
  const int n0   = blockIdx.y * 64;
  const int kb   = blockIdx.z * kChunk;
  const int ke   = (kb + kChunk < K) ? kb + kChunk : K;
  float* Cw = C + (size_t)blockIdx.z * cStride;

  v8f zero = {};
  v8f acc[NT];
  #pragma unroll
  for (int j = 0; j < NT; ++j) acc[j] = zero;

  for (int k0 = kb; k0 < ke; k0 += 32) {
    // ---- stage B tiles 64x32 bf16 (zero-padded weights) ----
    {
      const int n = tid >> 1;            // 0..63
      const int c = (tid & 1) * 16;      // 0 / 16
      const __bf16* src = Bt + (size_t)(n0 + n) * Kpad + k0 + c;
      __bf16* dstp = &Bs[n * LDSROW + c];
#ifdef ASYNC_LDS
      // gfx1250 GLOBAL_LOAD_ASYNC_TO_LDS_B128: per-lane 16B memory->LDS DMA,
      // tracked by ASYNCcnt (no VGPR round-trip for the weight tile).
      __builtin_amdgcn_global_load_async_to_lds_b128(
          (as1_v4i_t)src, (as3_v4i_t)dstp, 0, 0);
      __builtin_amdgcn_global_load_async_to_lds_b128(
          (as1_v4i_t)(src + 8), (as3_v4i_t)(dstp + 8), 0, 0);
#else
      uint4* dst = (uint4*)dstp;
      dst[0] = ((const uint4*)src)[0];
      dst[1] = ((const uint4*)src)[1];
#endif
    }
    // ---- stage A tile 64x32 (fp32 -> bf16), branch-free ----
    {
      const int r  = tid >> 1;           // 0..63
      const int ch = (tid & 1) * 16;     // 0 / 16
      const float* src = A + (size_t)(m0 + r) * lda + k0 + ch;
      float vals[16];
      #pragma unroll
      for (int j4 = 0; j4 < 4; ++j4) {
        float4 fv = ((const float4*)src)[j4];
        vals[j4*4+0] = fv.x; vals[j4*4+1] = fv.y;
        vals[j4*4+2] = fv.z; vals[j4*4+3] = fv.w;
      }
      if (k0 + ch + 16 > K) {            // K-tail: zero via selects, no loads
        #pragma unroll
        for (int j = 0; j < 16; ++j)
          vals[j] = (k0 + ch + j < K) ? vals[j] : 0.f;
      }
      __bf16* dst = &As[r * LDSROW + ch];
      #pragma unroll
      for (int j = 0; j < 16; ++j) dst[j] = (__bf16)vals[j];
      if (k0 + 32 < ke) __builtin_prefetch(src + 32, 0, 1);  // global_prefetch_b8
    }
#ifdef ASYNC_LDS
#  if __has_builtin(__builtin_amdgcn_s_wait_asynccnt)
    __builtin_amdgcn_s_wait_asynccnt(0);
#  else
    asm volatile("s_wait_asynccnt 0" ::: "memory");
#  endif
#endif
    __syncthreads();

    // ---- A fragment (shared across the 4 N-tiles) ----
    FragBF a;
    {
      const int row = (wave << 4) + (lane & 15);
      const int kh  = (lane >> 4) << 3;            // 0 or 8
      const __bf16* p = &As[row * LDSROW + kh];
      a.q[0] = *(const uint4*)p;                   // K 0..7  / 8..15
      a.q[1] = *(const uint4*)(p + 16);            // K 16..23/ 24..31
    }
    #pragma unroll
    for (int j = 0; j < NT; ++j) {
      FragBF b;
      const int n    = (j << 4) + (lane & 15);
      const int koff = (lane >> 4) << 3;           // 0 or 8
      const __bf16* p = &Bs[n * LDSROW + koff];
      b.q[0] = *(const uint4*)p;
      b.q[1] = *(const uint4*)(p + 16);
      acc[j] = __builtin_amdgcn_wmma_f32_16x16x32_bf16(
          false, a.v, false, b.v, (short)0, acc[j], false, false);
    }
    __syncthreads();
  }

  // ---- epilogue per C/D VGPR layout ----
  const int  rbase   = m0 + (wave << 4) + ((lane >> 4) << 3);
  const bool partial = (gridDim.z > 1);
  #pragma unroll
  for (int j = 0; j < NT; ++j) {
    const int col = n0 + (j << 4) + (lane & 15);
    if (col < N) {
      const float bval = (!partial && bias) ? bias[col] : 0.f;
      #pragma unroll
      for (int r = 0; r < 8; ++r) {
        float v = acc[j][r] + bval;
        if (!partial) v = apply_act<ACT>(v);
        Cw[(size_t)(rbase + r) * ldc + col] = v;
      }
    }
  }
}

// fixed-order split-K reduction (deterministic)
__global__ void reduce_sk(const float* __restrict__ part, const float* __restrict__ bias,
                          float* __restrict__ C, int MN, int N, int S, size_t stride)
{
  int idx = blockIdx.x * 256 + threadIdx.x;
  if (idx >= MN) return;
  float s = 0.f;
  for (int i = 0; i < S; ++i) s += part[idx + (size_t)i * stride];
  if (bias) s += bias[idx % N];
  C[idx] = s;
}

// =====================================================================
// Weight transpose + bf16 convert:  W[K][N] fp32  ->  Wt[Npad][Kpad] bf16
// (Npad multiple of 64 so the 64-wide N tiling never reads OOB)
// =====================================================================
__global__ void conv_wt_kernel(const float* __restrict__ W, __bf16* __restrict__ Wt,
                               int K, int N, int Kpad, int Npad)
{
  int idx = blockIdx.x * 256 + threadIdx.x;
  if (idx >= Kpad * Npad) return;
  int k = idx % Kpad, n = idx / Kpad;
  float v = (k < K && n < N) ? W[(size_t)k * N + n] : 0.f;
  Wt[idx] = (__bf16)v;
}

// ---------------- RevIN statistics: one wave per (b,v) series ----------------
__global__ __launch_bounds__(256) void revin_stats(const float* __restrict__ x,
                                                   float* __restrict__ mean,
                                                   float* __restrict__ stdev)
{
  int s = blockIdx.x * 8 + (threadIdx.x >> 5);     // series id 0..2047
  if (s >= BATCH * N_VARS) return;
  int lane = threadIdx.x & 31;
  int b = s >> 5, v = s & 31;
  const float* base = x + (size_t)b * SEQ_LEN * N_VARS + v;
  float sum = 0.f, sq = 0.f;
  for (int t = lane; t < SEQ_LEN; t += 32) {
    float f = base[(size_t)t * N_VARS];
    sum += f; sq += f * f;
  }
  for (int o = 16; o > 0; o >>= 1) {
    sum += __shfl_xor(sum, o, 32);
    sq  += __shfl_xor(sq,  o, 32);
  }
  if (lane == 0) {
    float mu  = sum * (1.f / SEQ_LEN);
    float var = sq * (1.f / SEQ_LEN) - mu * mu;
    mean[s]  = mu;
    stdev[s] = sqrtf(var + 1e-5f);
  }
}

// ---------------- patch gather (normalized, end-replicated, K padded to 32) ----
__global__ void build_patches(const float* __restrict__ x, const float* __restrict__ mean,
                              const float* __restrict__ stdev, float* __restrict__ P)
{
  int idx = blockIdx.x * 256 + threadIdx.x;
  if (idx >= T_TOK * 32) return;
  int j = idx & 31;
  int tok = idx >> 5;                 // (b*32+v)*64 + p
  if (j >= PATCH_LEN) { P[idx] = 0.f; return; }
  int p  = tok & 63;
  int bv = tok >> 6;
  int b = bv >> 5, v = bv & 31;
  int t = p * STRIDE_ + j; if (t > SEQ_LEN - 1) t = SEQ_LEN - 1;
  float f = x[((size_t)b * SEQ_LEN + t) * N_VARS + v];
  P[idx] = (f - mean[bv]) / stdev[bv];
}

// ---------------- split embed -> time path (+pos) & var path (transposed) ------
__global__ void split_paths(const float* __restrict__ emb, const float* __restrict__ W_pos,
                            float* __restrict__ enc_t, float* __restrict__ enc_v)
{
  int idx = blockIdx.x * 256 + threadIdx.x;
  if (idx >= T_TOK * D_MODEL) return;
  int d = idx & 127;
  int tok = idx >> 7;
  int p  = tok & 63;
  int bv = tok >> 6;
  int b = bv >> 5, v = bv & 31;
  float e = emb[idx];
  enc_t[idx] = e + W_pos[p];
  size_t vtok = ((size_t)(b * PATCH_NUM + p) * N_VARS + v);
  enc_v[vtok * D_MODEL + d] = e;
}

// ---------------- elementwise add (b may be null -> copy) ----------------
__global__ void addk(const float* __restrict__ a, const float* __restrict__ b,
                     float* __restrict__ out, int n)
{
  int idx = blockIdx.x * 256 + threadIdx.x;
  if (idx >= n) return;
  out[idx] = a[idx] + (b ? b[idx] : 0.f);
}

// ---------------- token norm: mode 0 = rmsnorm(w), 1 = layernorm(w,b) ---------
__global__ __launch_bounds__(256) void norm_kernel(
    const float* __restrict__ x, const float* __restrict__ w, const float* __restrict__ b,
    float* __restrict__ out, int T, int mode)
{
  int tok = blockIdx.x * 8 + (threadIdx.x >> 5);
  if (tok >= T) return;
  int lane = threadIdx.x & 31;
  const float* xr = x + (size_t)tok * D_MODEL;
  float v0 = xr[lane], v1 = xr[lane+32], v2 = xr[lane+64], v3 = xr[lane+96];
  float s  = v0+v1+v2+v3;
  float s2 = v0*v0+v1*v1+v2*v2+v3*v3;
  for (int o = 16; o > 0; o >>= 1) {
    s  += __shfl_xor(s,  o, 32);
    s2 += __shfl_xor(s2, o, 32);
  }
  float* orow = out + (size_t)tok * D_MODEL;
  if (mode == 0) {
    float inv = rsqrtf(s2 * (1.f/128.f) + 1e-5f);
    orow[lane]    = v0 * inv * w[lane];
    orow[lane+32] = v1 * inv * w[lane+32];
    orow[lane+64] = v2 * inv * w[lane+64];
    orow[lane+96] = v3 * inv * w[lane+96];
  } else {
    float mu  = s * (1.f/128.f);
    float var = s2 * (1.f/128.f) - mu * mu;
    float inv = rsqrtf(var + 1e-5f);
    orow[lane]    = (v0-mu)*inv*w[lane]    + b[lane];
    orow[lane+32] = (v1-mu)*inv*w[lane+32] + b[lane+32];
    orow[lane+64] = (v2-mu)*inv*w[lane+64] + b[lane+64];
    orow[lane+96] = (v3-mu)*inv*w[lane+96] + b[lane+96];
  }
}

// ---------------- causal depthwise conv (width 4) + SiLU, branch-free ----------
__global__ void conv_silu(const float* __restrict__ xz, const float* __restrict__ cw,
                          const float* __restrict__ cb, float* __restrict__ xc,
                          int L, int T)
{
  int idx = blockIdx.x * 256 + threadIdx.x;
  if (idx >= T * D_INNER) return;
  int c = idx & 255;
  int tok = idx >> 8;
  int t = tok & (L - 1);               // L is a power of two (64 / 32)
  float acc = cb[c];
  #pragma unroll
  for (int i = 0; i < 4; ++i) {
    int tt = t - 3 + i;
    int srctok = tok - 3 + i;
    srctok = srctok < 0 ? 0 : srctok;                       // clamped, valid addr
    float m = (tt >= 0) ? 1.f : 0.f;                        // masks the padding
    acc += m * xz[(size_t)srctok * 512 + c] * cw[i * D_INNER + c];
  }
  xc[idx] = act_silu(acc);
}

// ---------------- selective scan: block = one sequence, thread = channel -------
__global__ __launch_bounds__(256) void scan_kernel(
    const float* __restrict__ xc,     // [T,256] u
    const float* __restrict__ delta,  // [T,256]
    const float* __restrict__ dbc,    // [T,48]: B at 8..23, C at 24..39
    const float* __restrict__ A_log,  // [256,16]
    const float* __restrict__ Dp,     // [256]
    float* __restrict__ y,            // [T,256]
    int L)
{
  int seq = blockIdx.x;
  int d   = threadIdx.x;
  __shared__ float sB[16], sC[16];
  float Arow[16];
  #pragma unroll
  for (int n = 0; n < 16; ++n) Arow[n] = -__expf(A_log[d * 16 + n]);
  float Dd = Dp[d];
  float h[16];
  #pragma unroll
  for (int n = 0; n < 16; ++n) h[n] = 0.f;
  size_t base = (size_t)seq * L;
  for (int t = 0; t < L; ++t) {
    size_t tok = base + t;
    if (d < 32) {
      float v = dbc[tok * 48 + 8 + d];
      if (d < 16) sB[d] = v; else sC[d - 16] = v;
    }
    __syncthreads();
    float dl = delta[tok * 256 + d];
    float u  = xc[tok * 256 + d];
    float du = dl * u;
    float acc = 0.f;
    #pragma unroll
    for (int n = 0; n < 16; ++n) {
      float dA = __expf(dl * Arow[n]);
      h[n] = dA * h[n] + du * sB[n];
      acc += h[n] * sC[n];
    }
    y[tok * 256 + d] = acc + u * Dd;
    __syncthreads();
  }
}

// ---------------- y *= silu(z)  (z = xz cols 256..511) ----------------
__global__ void gate_kernel(float* __restrict__ y, const float* __restrict__ xz, int n)
{
  int idx = blockIdx.x * 256 + threadIdx.x;
  if (idx >= n) return;
  int c = idx & 255;
  int tok = idx >> 8;
  y[idx] *= act_silu(xz[(size_t)tok * 512 + 256 + c]);
}

// ---------------- concat time/var outputs -> [T,256] ----------------
__global__ void concat_tv(const float* __restrict__ out_t, const float* __restrict__ out_v,
                          float* __restrict__ cat)
{
  int idx = blockIdx.x * 256 + threadIdx.x;
  if (idx >= T_TOK * 256) return;
  int d = idx & 255;
  int tok = idx >> 8;                  // b,v,p order
  float val;
  if (d < 128) {
    val = out_t[(size_t)tok * 128 + d];
  } else {
    int p  = tok & 63;
    int bv = tok >> 6;
    int b = bv >> 5, v = bv & 31;
    size_t vtok = (size_t)(b * PATCH_NUM + p) * N_VARS + v;
    val = out_v[vtok * 128 + (d - 128)];
  }
  cat[idx] = val;
}

// ---------------- [b,v,p,d] -> flat [b*v][d*64+p] ----------------
__global__ void flatten_kernel(const float* __restrict__ src, float* __restrict__ flat)
{
  int idx = blockIdx.x * 256 + threadIdx.x;
  if (idx >= T_TOK * D_MODEL) return;
  int d = idx & 127;
  int tok = idx >> 7;
  int p  = tok & 63;
  int bv = tok >> 6;
  flat[(size_t)bv * (D_MODEL * PATCH_NUM) + d * PATCH_NUM + p] = src[idx];
}

// ---------------- de-normalize + transpose -> out[b,t,v] ----------------
__global__ void final_kernel(const float* __restrict__ hout, const float* __restrict__ mean,
                             const float* __restrict__ stdev, float* __restrict__ out)
{
  int idx = blockIdx.x * 256 + threadIdx.x;
  if (idx >= BATCH * PRED_LEN * N_VARS) return;
  int v = idx & 31;
  int rest = idx >> 5;
  int t = rest % PRED_LEN;
  int b = rest / PRED_LEN;
  int bv = b * N_VARS + v;
  out[idx] = hout[(size_t)bv * PRED_LEN + t] * stdev[bv] + mean[bv];
}

// =====================================================================
// Host-side orchestration
// =====================================================================
static inline void gemm(hipStream_t s, const float* A, int lda, const __bf16* Bt, int Kpad,
                        const float* bias, float* C, int ldc, int M, int N, int K, int act)
{
  dim3 g(CDIV(M, 64), CDIV(N, 64), 1);
  switch (act) {
    case ACT_SILU:
      gemm_wmma_bf16<ACT_SILU><<<g,128,0,s>>>(A,lda,Bt,Kpad,bias,C,ldc,N,K,K,0); break;
    case ACT_GELU:
      gemm_wmma_bf16<ACT_GELU><<<g,128,0,s>>>(A,lda,Bt,Kpad,bias,C,ldc,N,K,K,0); break;
    case ACT_SOFTPLUS:
      gemm_wmma_bf16<ACT_SOFTPLUS><<<g,128,0,s>>>(A,lda,Bt,Kpad,bias,C,ldc,N,K,K,0); break;
    default:
      gemm_wmma_bf16<ACT_NONE><<<g,128,0,s>>>(A,lda,Bt,Kpad,bias,C,ldc,N,K,K,0); break;
  }
}

static const __bf16* cvtW(hipStream_t s, const float* W, int K, int N, __bf16*& cur)
{
  int Kp = (K + 31) & ~31, Np = (N + 63) & ~63;
  int total = Kp * Np;
  conv_wt_kernel<<<CDIV(total, 256), 256, 0, s>>>(W, cur, K, N, Kp, Np);
  const __bf16* r = cur;
  cur += (size_t)total;
  return r;
}

struct BlkW { const __bf16 *inproj, *xproj, *dtw, *outproj, *fc1, *fc2; };

// one Mamba block + post-layernorm of the residual
static void run_layer(hipStream_t s, float* hid, float* res, bool first,
                      void* const* P, const BlkW& W,
                      const float* pnW, const float* pnB, int L,
                      float* t1, float* xz, float* xc, float* dbc,
                      float* delta, float* yb)
{
  const float* convw = (const float*)P[1];
  const float* convb = (const float*)P[2];
  const float* dtb   = (const float*)P[5];
  const float* alog  = (const float*)P[6];
  const float* Dp    = (const float*)P[7];
  const float* nw1   = (const float*)P[9];
  const float* b1    = (const float*)P[11];
  const float* b2    = (const float*)P[13];
  const float* nw2   = (const float*)P[14];
  const int T = T_TOK;

  addk<<<CDIV(T*128,256),256,0,s>>>(hid, first ? nullptr : res, res, T*128);
  norm_kernel<<<CDIV(T,8),256,0,s>>>(res, nw1, nullptr, t1, T, 0);
  gemm(s, t1, 128, W.inproj, 128, nullptr, xz, 512, T, 512, 128, ACT_NONE);
  conv_silu<<<CDIV(T*256,256),256,0,s>>>(xz, convw, convb, xc, L, T);
  gemm(s, xc, 256, W.xproj, 256, nullptr, dbc, 48, T, DT_RANK + 2*D_STATE, 256, ACT_NONE);
  gemm(s, dbc, 48, W.dtw, 32, dtb, delta, 256, T, 256, DT_RANK, ACT_SOFTPLUS);
  scan_kernel<<<T / L, 256, 0, s>>>(xc, delta, dbc, alog, Dp, yb, L);
  gate_kernel<<<CDIV(T*256,256),256,0,s>>>(yb, xz, T*256);
  gemm(s, yb, 256, W.outproj, 256, nullptr, t1, 128, T, 128, 256, ACT_NONE);
  addk<<<CDIV(T*128,256),256,0,s>>>(t1, res, res, T*128);
  norm_kernel<<<CDIV(T,8),256,0,s>>>(res, nw2, nullptr, t1, T, 0);
  gemm(s, t1, 128, W.fc1, 128, b1, yb, 256, T, D_FF, 128, ACT_GELU);
  gemm(s, yb, 256, W.fc2, 256, b2, hid, 128, T, 128, D_FF, ACT_NONE);
  norm_kernel<<<CDIV(T,8),256,0,s>>>(res, pnW, pnB, res, T, 1);
}

static void run_path(hipStream_t s, float* enc, float* outbuf, int L,
                     void* const* d_in, int base0, int base1,
                     const BlkW& W0, const BlkW& W1,
                     const float* pnW, const float* pnB,
                     float* res, float* t1, float* xz, float* xc,
                     float* dbc, float* delta, float* yb)
{
  run_layer(s, enc, res, true,  d_in + base0, W0, pnW, pnB, L, t1, xz, xc, dbc, delta, yb);
  run_layer(s, enc, res, false, d_in + base1, W1, pnW, pnB, L, t1, xz, xc, dbc, delta, yb);
  addk<<<CDIV(T_TOK*128,256),256,0,s>>>(enc, res, t1, T_TOK*128);
  norm_kernel<<<CDIV(T_TOK,8),256,0,s>>>(t1, pnW, pnB, outbuf, T_TOK, 1);
}

extern "C" void kernel_launch(void* const* d_in, const int* in_sizes, int n_in,
                              void* d_out, int out_size, void* d_ws, size_t ws_size,
                              hipStream_t stream)
{
  (void)in_sizes; (void)n_in; (void)out_size; (void)ws_size;
  const float* x_enc  = (const float*)d_in[0];
  const float* W_P_w  = (const float*)d_in[1];
  const float* W_P_b  = (const float*)d_in[2];
  const float* W_pos  = (const float*)d_in[3];
  const float* norm_w = (const float*)d_in[4];
  const float* norm_b = (const float*)d_in[5];
  const float* nvw    = (const float*)d_in[6];
  const float* nvb    = (const float*)d_in[7];
  const float* ff_w1  = (const float*)d_in[8];
  const float* ff_b1  = (const float*)d_in[9];
  const float* ff_w2  = (const float*)d_in[10];
  const float* ff_b2  = (const float*)d_in[11];
  const float* head_w = (const float*)d_in[12];
  const float* head_b = (const float*)d_in[13];
  const int pbase[4] = {14, 29, 44, 59};   // time_p0, time_p1, var_p0, var_p1

  // ---- carve workspace ----
  const size_t T = T_TOK;
  float* f = (float*)d_ws;
  float* mean    = f; f += 2048;
  float* stdev   = f; f += 2048;
  float* patches = f; f += T * 32;
  float* enc_t   = f; f += T * 128;
  float* enc_v   = f; f += T * 128;
  float* out_t   = f; f += T * 128;
  float* out_v   = f; f += T * 128;
  float* res     = f; f += T * 128;
  float* t1      = f; f += T * 128;
  float* xz      = f; f += T * 512;
  float* xc      = f; f += T * 256;
  float* dbc     = f; f += T * 48;
  float* delta   = f; f += T * 256;
  float* yb      = f; f += T * 256;
  float* cat     = f; f += T * 256;
  float* ffh     = f; f += T * 128;
  float* flat    = f; f += T * 128;      // == 2048 x 8192
  float* hout    = f; f += 2048 * PRED_LEN;
  float* skbuf   = f; f += (size_t)8 * 2048 * PRED_LEN;   // split-K partials
  __bf16* wcur   = (__bf16*)f;

  // ---- convert + transpose all weights to bf16 [Npad][Kpad] ----
  const __bf16* wpT = cvtW(stream, W_P_w, PATCH_LEN, D_MODEL, wcur);
  BlkW BW[4];
  for (int i = 0; i < 4; ++i) {
    void* const* P = d_in + pbase[i];
    BW[i].inproj  = cvtW(stream, (const float*)P[0],  D_MODEL, 2*D_INNER, wcur);
    BW[i].xproj   = cvtW(stream, (const float*)P[3],  D_INNER, DT_RANK + 2*D_STATE, wcur);
    BW[i].dtw     = cvtW(stream, (const float*)P[4],  DT_RANK, D_INNER, wcur);
    BW[i].outproj = cvtW(stream, (const float*)P[8],  D_INNER, D_MODEL, wcur);
    BW[i].fc1     = cvtW(stream, (const float*)P[10], D_MODEL, D_FF, wcur);
    BW[i].fc2     = cvtW(stream, (const float*)P[12], D_FF, D_MODEL, wcur);
  }
  const __bf16* ffw1T = cvtW(stream, ff_w1, 2*D_MODEL, D_MODEL, wcur);
  const __bf16* ffw2T = cvtW(stream, ff_w2, D_MODEL, D_MODEL, wcur);
  const __bf16* headT = cvtW(stream, head_w, D_MODEL * PATCH_NUM, PRED_LEN, wcur);

  // ---- RevIN stats + patch embedding ----
  revin_stats<<<CDIV(2048,8),256,0,stream>>>(x_enc, mean, stdev);
  build_patches<<<CDIV((int)(T*32),256),256,0,stream>>>(x_enc, mean, stdev, patches);
  gemm(stream, patches, 32, wpT, 32, W_P_b, t1, 128, (int)T, D_MODEL, PATCH_LEN, ACT_NONE);
  split_paths<<<CDIV((int)(T*128),256),256,0,stream>>>(t1, W_pos, enc_t, enc_v);

  // ---- time path (L=64) and variable path (L=32) ----
  run_path(stream, enc_t, out_t, PATCH_NUM, d_in, pbase[0], pbase[1],
           BW[0], BW[1], norm_w, norm_b, res, t1, xz, xc, dbc, delta, yb);
  run_path(stream, enc_v, out_v, N_VARS, d_in, pbase[2], pbase[3],
           BW[2], BW[3], nvw, nvb, res, t1, xz, xc, dbc, delta, yb);

  // ---- fusion FF ----
  concat_tv<<<CDIV((int)(T*256),256),256,0,stream>>>(out_t, out_v, cat);
  gemm(stream, cat, 256, ffw1T, 256, ff_b1, ffh, 128, (int)T, D_MODEL, 2*D_MODEL, ACT_GELU);
  gemm(stream, ffh, 128, ffw2T, 128, ff_b2, t1, 128, (int)T, D_MODEL, D_MODEL, ACT_NONE);
  flatten_kernel<<<CDIV((int)(T*128),256),256,0,stream>>>(t1, flat);

  // ---- head: split-K (8 deterministic partials) then fixed-order reduce ----
  {
    const int M = BATCH * N_VARS, N = PRED_LEN, K = D_MODEL * PATCH_NUM, S = 8;
    dim3 g(CDIV(M, 64), CDIV(N, 64), S);
    gemm_wmma_bf16<ACT_NONE><<<g, 128, 0, stream>>>(flat, K, headT, K, nullptr, skbuf, N,
                                                    N, K, K / S, (size_t)M * N);
    reduce_sk<<<CDIV(M * N, 256), 256, 0, stream>>>(skbuf, head_b, hout,
                                                    M * N, N, S, (size_t)M * N);
  }
  final_kernel<<<CDIV(BATCH*PRED_LEN*N_VARS,256),256,0,stream>>>(
      hout, mean, stdev, (float*)d_out);
}
